// simpleGCN2_35656818491447
// MI455X (gfx1250) — matrix operand order repaired and checked by reading, not compiled, not run on hardware
//
#include <hip/hip_runtime.h>
#include <hip/hip_bf16.h>

#define NN      160000
#define EE      1280000
#define CC      64
#define GENES   5000
#define NGRAPH  32

typedef __attribute__((ext_vector_type(16))) __bf16 v16bf;
typedef __attribute__((ext_vector_type(8)))  float  v8f;

// ---------------------------------------------------------------------------
// Degree / normalization
// ---------------------------------------------------------------------------
__global__ void k_deg_init(float* __restrict__ deg) {
    int n = blockIdx.x * blockDim.x + threadIdx.x;
    if (n < NN) deg[n] = 1.0f;                 // self loop
}

__global__ void k_deg_count(const int* __restrict__ dst, float* __restrict__ deg) {
    int e = blockIdx.x * blockDim.x + threadIdx.x;
    if (e < EE) atomicAdd(&deg[dst[e]], 1.0f);
}

__global__ void k_deg_rsqrt(float* __restrict__ deg) {
    int n = blockIdx.x * blockDim.x + threadIdx.x;
    if (n < NN) {
        float d = deg[n];
        deg[n] = (d > 0.0f) ? rsqrtf(d) : 0.0f;
    }
}

// ---------------------------------------------------------------------------
// WMMA fragment helpers (bf16 A/B, f32 accum), layouts per CDNA5 ISA 7.12.2
// ---------------------------------------------------------------------------
// B[k][n] = W[c][k]  (computes h @ W^T).  8 VGPRs: lane%16 = column n,
// lane/16==0 -> K {0..7,16..23}, ==1 -> {8..15,24..31} (same as A).
__device__ __forceinline__ v16bf load_B_frag(const float* __restrict__ W,
                                             int lane, int ctile, int kstep) {
    int c     = ctile * 16 + (lane & 15);
    int kbase = kstep * 32 + (lane >> 4) * 8;
    const float* wrow = W + c * CC + kbase;
    v16bf b;
#pragma unroll
    for (int j = 0; j < 16; ++j) b[j] = (__bf16)wrow[j + (j >> 3) * 8];
    return b;
}

// A 16x32 bf16: lane%16 = row; lane/16==0 -> K {0..7,16..23}, ==1 -> {8..15,24..31}
__device__ __forceinline__ v16bf load_A_frag(const float* __restrict__ hrow,
                                             int lane, int kstep) {
    int half  = lane >> 4;
    int base0 = kstep * 32 + half * 8;
    int base1 = base0 + 16;
    v16bf a;
#pragma unroll
    for (int j = 0; j < 8; ++j) a[j]     = (__bf16)hrow[base0 + j];
#pragma unroll
    for (int j = 0; j < 8; ++j) a[8 + j] = (__bf16)hrow[base1 + j];
    return a;
}

// Same A fragment but h0 generated on the fly: h0[r][k] = exp_w[gene][k]*x + eb
__device__ __forceinline__ v16bf load_A_frag_embed(const float* __restrict__ erow,
                                                   float xv, float eb,
                                                   int lane, int kstep) {
    int half  = lane >> 4;
    int base0 = kstep * 32 + half * 8;
    int base1 = base0 + 16;
    v16bf a;
#pragma unroll
    for (int j = 0; j < 8; ++j) a[j]     = (__bf16)(erow[base0 + j] * xv + eb);
#pragma unroll
    for (int j = 0; j < 8; ++j) a[8 + j] = (__bf16)(erow[base1 + j] * xv + eb);
    return a;
}

__device__ __forceinline__ void wmma_row_tile(v16bf a0, v16bf a1,
                                              const float* __restrict__ W,
                                              int lane, int rowBase,
                                              float* __restrict__ T) {
    int rof = (lane >> 4) * 8;
    int col = lane & 15;
#pragma unroll
    for (int ct = 0; ct < 4; ++ct) {
        v8f acc = {};
        v16bf b0 = load_B_frag(W, lane, ct, 0);
        acc = __builtin_amdgcn_wmma_f32_16x16x32_bf16(false, a0, false, b0,
                                                      (short)0, acc, false, false);
        v16bf b1 = load_B_frag(W, lane, ct, 1);
        acc = __builtin_amdgcn_wmma_f32_16x16x32_bf16(false, a1, false, b1,
                                                      (short)0, acc, false, false);
#pragma unroll
        for (int j = 0; j < 8; ++j)
            T[(size_t)(rowBase + rof + j) * CC + ct * 16 + col] = acc[j];
    }
}

// GEMM1 with fused embedding:  T = (exp_w[gene]*x + exp_b[gene]) @ w1^T
__global__ __launch_bounds__(256)
void k_gemm_embed(const float* __restrict__ x, const float* __restrict__ exp_w,
                  const float* __restrict__ exp_b, const float* __restrict__ W,
                  float* __restrict__ T) {
    int lane    = threadIdx.x & 31;
    int wave    = threadIdx.x >> 5;
    int rowBase = (blockIdx.x * 8 + wave) * 16;   // grid exactly tiles NN
    int r       = rowBase + (lane & 15);
    int gene    = r % GENES;
    float xv = x[r];
    float eb = exp_b[gene];
    const float* erow = exp_w + (size_t)gene * CC;
    v16bf a0 = load_A_frag_embed(erow, xv, eb, lane, 0);
    v16bf a1 = load_A_frag_embed(erow, xv, eb, lane, 1);
    wmma_row_tile(a0, a1, W, lane, rowBase, T);
}

// Plain GEMM:  T = H @ W^T
__global__ __launch_bounds__(256)
void k_gemm(const float* __restrict__ H, const float* __restrict__ W,
            float* __restrict__ T) {
    int lane    = threadIdx.x & 31;
    int wave    = threadIdx.x >> 5;
    int rowBase = (blockIdx.x * 8 + wave) * 16;
    const float* hrow = H + (size_t)(rowBase + (lane & 15)) * CC;
    v16bf a0 = load_A_frag(hrow, lane, 0);
    v16bf a1 = load_A_frag(hrow, lane, 1);
    wmma_row_tile(a0, a1, W, lane, rowBase, T);
}

// ---------------------------------------------------------------------------
// GCN aggregation
// ---------------------------------------------------------------------------
// Acc[n][c] = dinv[n]^2 * T[n][c]   (self-loop term)
__global__ void k_self_init(const float* __restrict__ dinv,
                            const float* __restrict__ T,
                            float* __restrict__ Acc) {
    size_t i = (size_t)blockIdx.x * blockDim.x + threadIdx.x;
    if (i < (size_t)NN * CC) {
        float di = dinv[i >> 6];
        Acc[i] = di * di * T[i];
    }
}

// Edge-parallel scatter: each thread does one edge x 4 channels (float4 gather,
// 4 f32 atomics landing in L2 since the 41MB feature map is L2-resident).
__global__ void k_edge_scatter(const int* __restrict__ src,
                               const int* __restrict__ dst,
                               const float* __restrict__ dinv,
                               const float* __restrict__ T,
                               float* __restrict__ Acc) {
    size_t t = (size_t)blockIdx.x * blockDim.x + threadIdx.x;
    if (t >= (size_t)EE * 16) return;
    int e  = (int)(t >> 4);
    int cq = (int)(t & 15) * 4;
    int s = src[e], d = dst[e];
    float norm = dinv[s] * dinv[d];
    const float4 v = *(const float4*)(T + (size_t)s * CC + cq);
    float* o = Acc + (size_t)d * CC + cq;
    atomicAdd(o + 0, norm * v.x);
    atomicAdd(o + 1, norm * v.y);
    atomicAdd(o + 2, norm * v.z);
    atomicAdd(o + 3, norm * v.w);
}

// H[n][c] = tanh(Acc[n][c] + b[c])
__global__ void k_bias_tanh(const float* __restrict__ Acc,
                            const float* __restrict__ bias,
                            float* __restrict__ H) {
    size_t i = (size_t)blockIdx.x * blockDim.x + threadIdx.x;
    if (i < (size_t)NN * CC) {
        H[i] = tanhf(Acc[i] + bias[i & 63]);
    }
}

// ---------------------------------------------------------------------------
// Pool + head
// ---------------------------------------------------------------------------
__global__ __launch_bounds__(256)
void k_pool(const float* __restrict__ H, float* __restrict__ pooled) {
    __shared__ float sh[256];
    int g = blockIdx.x;
    int c = threadIdx.x & 63;
    int chunk = threadIdx.x >> 6;                 // 0..3, 1250 rows each
    const float* base = H + (size_t)g * GENES * CC + c;
    float s = 0.0f;
    for (int i = chunk * 1250; i < (chunk + 1) * 1250; ++i)
        s += base[(size_t)i * CC];
    sh[threadIdx.x] = s;
    __syncthreads();
    if (chunk == 0) {
        float tot = sh[c] + sh[c + 64] + sh[c + 128] + sh[c + 192];
        pooled[g * CC + c] = tot * (1.0f / GENES);
    }
}

__global__ void k_head(const float* __restrict__ pooled,
                       const float* __restrict__ lin_w,
                       const float* __restrict__ lin_b,
                       float* __restrict__ out) {
    int i = blockIdx.x * blockDim.x + threadIdx.x;
    if (i >= NGRAPH * CC) return;
    int g = i >> 6, c = i & 63;
    float s = lin_b[c];
    const float* p = pooled + g * CC;
    const float* w = lin_w + c * CC;
#pragma unroll 8
    for (int k = 0; k < CC; ++k) s += p[k] * w[k];
    out[i] = s;
}

// ---------------------------------------------------------------------------
extern "C" void kernel_launch(void* const* d_in, const int* in_sizes, int n_in,
                              void* d_out, int out_size, void* d_ws, size_t ws_size,
                              hipStream_t stream) {
    const float* x      = (const float*)d_in[0];
    const int*   eidx   = (const int*)d_in[1];     // [2, E]
    /* d_in[2] = batch: layout known (NUM_GENES consecutive per graph), unused */
    const float* exp_w  = (const float*)d_in[3];
    const float* exp_b  = (const float*)d_in[4];
    const float* w1     = (const float*)d_in[5];
    const float* b1     = (const float*)d_in[6];
    const float* w2     = (const float*)d_in[7];
    const float* b2     = (const float*)d_in[8];
    const float* lin_w  = (const float*)d_in[9];
    const float* lin_b  = (const float*)d_in[10];
    float*       out    = (float*)d_out;

    const int* src = eidx;
    const int* dst = eidx + EE;

    // workspace layout: dinv | bufA | bufB | pooled  (~83 MB total)
    char* ws = (char*)d_ws;
    float* dinv   = (float*)ws;                                 // NN
    float* bufA   = (float*)(ws + (size_t)NN * 4);              // NN*CC
    float* bufB   = (float*)(ws + (size_t)NN * 4 + (size_t)NN * CC * 4);
    float* pooled = (float*)(ws + (size_t)NN * 4 + 2 * (size_t)NN * CC * 4);

    const int TB = 256;
    const int gridNode = (NN + TB - 1) / TB;                    // 625
    const int gridEdge = (EE + TB - 1) / TB;                    // 5000
    const int gridNC   = (int)(((size_t)NN * CC + TB - 1) / TB);// 40000
    const int gridScat = (int)(((size_t)EE * 16 + TB - 1) / TB);// 80000
    const int gridGemm = NN / (8 * 16);                         // 1250 (exact)

    // symmetric normalization
    k_deg_init <<<gridNode, TB, 0, stream>>>(dinv);
    k_deg_count<<<gridEdge, TB, 0, stream>>>(dst, dinv);
    k_deg_rsqrt<<<gridNode, TB, 0, stream>>>(dinv);

    // conv1: fused embed + WMMA GEMM -> bufA; aggregate -> bufB; tanh -> bufA
    k_gemm_embed  <<<gridGemm, TB, 0, stream>>>(x, exp_w, exp_b, w1, bufA);
    k_self_init   <<<gridNC,   TB, 0, stream>>>(dinv, bufA, bufB);
    k_edge_scatter<<<gridScat, TB, 0, stream>>>(src, dst, dinv, bufA, bufB);
    k_bias_tanh   <<<gridNC,   TB, 0, stream>>>(bufB, b1, bufA);

    // conv2: WMMA GEMM -> bufB; aggregate -> bufA; tanh -> bufB
    k_gemm        <<<gridGemm, TB, 0, stream>>>(bufA, w2, bufB);
    k_self_init   <<<gridNC,   TB, 0, stream>>>(dinv, bufB, bufA);
    k_edge_scatter<<<gridScat, TB, 0, stream>>>(src, dst, dinv, bufB, bufA);
    k_bias_tanh   <<<gridNC,   TB, 0, stream>>>(bufA, b2, bufB);

    // mean pool + linear head
    k_pool<<<NGRAPH, TB, 0, stream>>>(bufB, pooled);
    k_head<<<(NGRAPH * CC + TB - 1) / TB, TB, 0, stream>>>(pooled, lin_w, lin_b, out);
}
